// NLDS_18863496364132
// MI455X (gfx1250) — matrix operand order repaired and checked by reading, not compiled, not run on hardware
//
#include <hip/hip_runtime.h>

// ---------------------------------------------------------------------------
// Diagonal LTI scan + fused projection for MI455X (gfx1250, wave32, WMMA)
//   h_t = A * h_{t-1} + x_t            (per-channel scan, fp32)
//   out = hs @ BC                      (GEMM via v_wmma_f32_16x16x32_f16,
//                                       async global->LDS double buffering)
// ---------------------------------------------------------------------------

typedef __attribute__((ext_vector_type(16))) _Float16 v16h;
typedef __attribute__((ext_vector_type(8)))  _Float16 v8h;
typedef __attribute__((ext_vector_type(8)))  float    v8f;

// Matches the builtin's parameter type exactly (per hipcc diagnostic):
// 'int __attribute__((__vector_size__(4*sizeof(int))))' pointers in AS(1)/AS(3).
typedef int nlds_v4i __attribute__((vector_size(16)));
typedef __attribute__((address_space(1))) nlds_v4i* nlds_gptr;
typedef __attribute__((address_space(3))) nlds_v4i* nlds_lptr;

#define NLDS_HAS_ASYNC (__has_builtin(__builtin_amdgcn_global_load_async_to_lds_b128) && \
                        __has_builtin(__builtin_amdgcn_s_wait_asynccnt))

namespace nlds {
constexpr int B = 8;
constexpr int S = 2048;
constexpr int D = 1024;   // K of the GEMM
constexpr int O = 1024;   // N of the GEMM
constexpr int M = B * S;  // 16384 rows

constexpr int BM  = 128;    // workgroup tile M
constexpr int BN  = 128;    // workgroup tile N
constexpr int BK  = 32;     // K step per WMMA
constexpr int LDK = BK + 8; // padded LDS row (halfs) -> conflict-free frags
constexpr int NK  = D / BK; // 32 K-iterations
}

// ---------------------------------------------------------------------------
// Kernel 1: BC[D][O] fp32  ->  Bt[O][D] f16  (transpose + downconvert)
// ---------------------------------------------------------------------------
__global__ __launch_bounds__(256) void nlds_bct(const float* __restrict__ BC,
                                                _Float16* __restrict__ Bt) {
  using namespace nlds;
  __shared__ float sh[32][33];
  const int tx = threadIdx.x & 31;
  const int ty = threadIdx.x >> 5;  // 0..7
  const int d0 = blockIdx.x * 32;
  const int o0 = blockIdx.y * 32;
#pragma unroll
  for (int i = 0; i < 32; i += 8)
    sh[ty + i][tx] = BC[(size_t)(d0 + ty + i) * O + (o0 + tx)];
  __syncthreads();
#pragma unroll
  for (int i = 0; i < 32; i += 8)
    Bt[(size_t)(o0 + ty + i) * D + (d0 + tx)] = (_Float16)sh[tx][ty + i];
}

// ---------------------------------------------------------------------------
// Kernel 2: sequential scan, one thread per (b, d) channel.
// x is [B][S][D]; hs output is f16 [M][D] with m = b*S + s (same linear index).
// Lanes cover consecutive d -> fully coalesced 128B loads / 64B f16 stores.
// ---------------------------------------------------------------------------
__global__ __launch_bounds__(256) void nlds_scan(const float* __restrict__ x,
                                                 const float* __restrict__ A,
                                                 _Float16* __restrict__ hs) {
  using namespace nlds;
  const int tid = blockIdx.x * 256 + threadIdx.x;  // 0 .. B*D-1
  const int b = tid >> 10;          // / D
  const int d = tid & (D - 1);
  const float a = A[d];
  const float* xp = x + (size_t)b * S * D + d;
  _Float16*   hp = hs + (size_t)b * S * D + d;
  float h = 0.0f;
  for (int s = 0; s < S; s += 4) {
    // pull the x-stream ahead of the dependent FMA chain
    __builtin_prefetch((const void*)(xp + 16 * D), 0, 0);
    const float x0 = xp[0 * D];
    const float x1 = xp[1 * D];
    const float x2 = xp[2 * D];
    const float x3 = xp[3 * D];
    h = fmaf(h, a, x0); hp[0 * D] = (_Float16)h;
    h = fmaf(h, a, x1); hp[1 * D] = (_Float16)h;
    h = fmaf(h, a, x2); hp[2 * D] = (_Float16)h;
    h = fmaf(h, a, x3); hp[3 * D] = (_Float16)h;
    xp += 4 * D;
    hp += 4 * D;
  }
}

// ---------------------------------------------------------------------------
// Kernel 3: TN GEMM  out[M][O] (fp32) = hs[M][D] (f16) * Bt[O][D]^T (f16)
// 8 waves / block in a 4(M) x 2(N) grid; each wave: 2x4 tiles of 16x16,
// v_wmma_f32_16x16x32_f16 with fp32 accumulators. Tiles staged with
// GLOBAL_LOAD_ASYNC_TO_LDS_B128 into a double buffer (ASYNCcnt-tracked).
// ---------------------------------------------------------------------------
__global__ __launch_bounds__(256) void nlds_gemm(const _Float16* __restrict__ Ahs,
                                                 const _Float16* __restrict__ Bt,
                                                 float* __restrict__ out) {
  using namespace nlds;
  __shared__ __align__(16) _Float16 sA[2][BM * LDK];
  __shared__ __align__(16) _Float16 sB[2][BN * LDK];

  const int tid  = threadIdx.x;
  const int lane = tid & 31;
  const int wave = tid >> 5;
  const int wm   = wave & 3;   // M-offset = wm*32
  const int wn   = wave >> 2;  // N-offset = wn*64
  const int m0   = blockIdx.x * BM;
  const int n0   = blockIdx.y * BN;

  v8f acc[2][4] = {};

  // staging: 512 chunks of 8 halfs (16B) per tile, 2 chunks per thread
  const int c0 = tid, c1 = tid + 256;
  const int ar0 = c0 >> 2, ak0 = (c0 & 3) * 8;
  const int ar1 = c1 >> 2, ak1 = (c1 & 3) * 8;

  const _Float16* gA0 = &Ahs[(size_t)(m0 + ar0) * D + ak0];
  const _Float16* gA1 = &Ahs[(size_t)(m0 + ar1) * D + ak1];
  const _Float16* gB0 = &Bt [(size_t)(n0 + ar0) * D + ak0];
  const _Float16* gB1 = &Bt [(size_t)(n0 + ar1) * D + ak1];

  // CDNA5 WMMA f16 fragment lane layout (ISA 7.12.2):
  //   A 16x32: lanes 0-15 -> K 0..7 & 16..23 ; lanes 16-31 -> K 8..15 & 24..31
  //   B 32x16: lanes 0-15 -> K 0..7 & 16..23 ; lanes 16-31 -> K 8..15 & 24..31
  const int kA = (lane < 16) ? 0 : 8;
  const int kB = (lane < 16) ? 0 : 8;
  const int ml = lane & 15;

#if NLDS_HAS_ASYNC
  auto stage_async = [&](int buf, int kt) {
    __builtin_amdgcn_global_load_async_to_lds_b128(
        (nlds_gptr)(gA0 + kt), (nlds_lptr)&sA[buf][ar0 * LDK + ak0], 0, 0);
    __builtin_amdgcn_global_load_async_to_lds_b128(
        (nlds_gptr)(gA1 + kt), (nlds_lptr)&sA[buf][ar1 * LDK + ak1], 0, 0);
    __builtin_amdgcn_global_load_async_to_lds_b128(
        (nlds_gptr)(gB0 + kt), (nlds_lptr)&sB[buf][ar0 * LDK + ak0], 0, 0);
    __builtin_amdgcn_global_load_async_to_lds_b128(
        (nlds_gptr)(gB1 + kt), (nlds_lptr)&sB[buf][ar1 * LDK + ak1], 0, 0);
  };
  stage_async(0, 0);
#endif

  // unroll 2: 'buf' stays compile-time constant (double buffer period) without
  // 32x body duplication / per-iteration address-table blowup.
#pragma unroll 2
  for (int i = 0; i < NK; ++i) {
#if NLDS_HAS_ASYNC
    const int buf = i & 1;
    if (i + 1 < NK) {
      stage_async(buf ^ 1, (i + 1) * BK);       // background prefetch
      __builtin_amdgcn_s_wait_asynccnt(4);      // this wave's buf-ops done
    } else {
      __builtin_amdgcn_s_wait_asynccnt(0);
    }
#else
    const int buf = 0;
    const int kt = i * BK;
    *(float4*)&sA[0][ar0 * LDK + ak0] = *(const float4*)(gA0 + kt);
    *(float4*)&sA[0][ar1 * LDK + ak1] = *(const float4*)(gA1 + kt);
    *(float4*)&sB[0][ar0 * LDK + ak0] = *(const float4*)(gB0 + kt);
    *(float4*)&sB[0][ar1 * LDK + ak1] = *(const float4*)(gB1 + kt);
#endif
    __syncthreads();  // all waves' tile data visible in LDS

    union V16 { v16h v; v8h h[2]; };
    V16 af[2], bf[4];
#pragma unroll
    for (int mi = 0; mi < 2; ++mi) {
      const _Float16* p = &sA[buf][(wm * 32 + mi * 16 + ml) * LDK + kA];
      af[mi].h[0] = *(const v8h*)(p);
      af[mi].h[1] = *(const v8h*)(p + 16);
    }
#pragma unroll
    for (int ni = 0; ni < 4; ++ni) {
      const _Float16* p = &sB[buf][(wn * 64 + ni * 16 + ml) * LDK + kB];
      bf[ni].h[0] = *(const v8h*)(p);
      bf[ni].h[1] = *(const v8h*)(p + 16);
    }

#pragma unroll
    for (int mi = 0; mi < 2; ++mi)
#pragma unroll
      for (int ni = 0; ni < 4; ++ni)
        acc[mi][ni] = __builtin_amdgcn_wmma_f32_16x16x32_f16(
            false, af[mi].v, false, bf[ni].v, (short)0, acc[mi][ni],
            false, false);

    __syncthreads();  // all waves done reading buf before it is overwritten
  }

  // C/D layout: VGPR r -> row r (lanes 0-15) / r+8 (lanes 16-31), col = lane%16
  // Non-temporal: out is write-once, keep it out of the 192MB L2.
  const int rbase = (lane < 16) ? 0 : 8;
#pragma unroll
  for (int mi = 0; mi < 2; ++mi)
#pragma unroll
    for (int ni = 0; ni < 4; ++ni)
#pragma unroll
      for (int r = 0; r < 8; ++r) {
        const int m = m0 + wm * 32 + mi * 16 + rbase + r;
        const int n = n0 + wn * 64 + ni * 16 + ml;
        __builtin_nontemporal_store(acc[mi][ni][r], &out[(size_t)m * O + n]);
      }
}

// ---------------------------------------------------------------------------
extern "C" void kernel_launch(void* const* d_in, const int* in_sizes, int n_in,
                              void* d_out, int out_size, void* d_ws, size_t ws_size,
                              hipStream_t stream) {
  using namespace nlds;
  const float* x  = (const float*)d_in[0];  // [B,S,D] fp32
  const float* A  = (const float*)d_in[1];  // [D]     fp32
  const float* BC = (const float*)d_in[2];  // [D,O]   fp32
  float* out = (float*)d_out;               // [B,S,O] fp32

  // workspace: [ Bt f16 : O*D ][ hs f16 : M*D ]  (2 MiB + 32 MiB)
  _Float16* Bt = (_Float16*)d_ws;
  _Float16* hs = (_Float16*)((char*)d_ws + (size_t)O * D * sizeof(_Float16));

  nlds_bct<<<dim3(D / 32, O / 32), 256, 0, stream>>>(BC, Bt);
  nlds_scan<<<(B * D) / 256, 256, 0, stream>>>(x, A, hs);
  nlds_gemm<<<dim3(M / BM, O / BN), 256, 0, stream>>>(hs, Bt, out);
}
